// Multi_Granularity_Align_One_Level_62886911148617
// MI455X (gfx1250) — matrix operand order repaired and checked by reading, not compiled, not run on hardware
//
#include <hip/hip_runtime.h>
#include <hip/hip_bf16.h>

// ---------------------------------------------------------------------------
// Types for CDNA5 WMMA (wave32): bf16 A/B fragments (16 x bf16 = 8 VGPRs),
// fp32 C/D accumulator (8 floats = 8 VGPRs).
// ---------------------------------------------------------------------------
typedef __attribute__((ext_vector_type(16))) __bf16 v16bf_t;
typedef __attribute__((ext_vector_type(8)))  float  v8f;

union BF16x16 {
    v16bf_t        v;
    unsigned short u[16];
    uint4          q[2];
};

__device__ __forceinline__ unsigned short f2bf(float f) {
    unsigned int x = __float_as_uint(f);
    x += 0x7FFFu + ((x >> 16) & 1u);          // round-to-nearest-even
    return (unsigned short)(x >> 16);
}
__device__ __forceinline__ float bf2f(unsigned short s) {
    return __uint_as_float(((unsigned int)s) << 16);
}

#define HH 128
#define WW 128
#define HWSZ 16384      // 128*128
#define N4 8            // 4 branches * B(=2)
#define ALIGN_SZ 2097152   // B*64*HW floats per aligned output region
#define OFF_BASE 10485760  // 5*ALIGN_SZ : start of off_feat region in d_out

// ---------------------------------------------------------------------------
// Weight packing: fp32 [Cout][Cin][3][3] -> bf16 A-fragments
// layout [r][kc][mt][lane][16]; lane half h holds k = 16h + i (contiguous, same order as B).
// ---------------------------------------------------------------------------
__global__ void pack_w_kernel(const float* __restrict__ W, unsigned short* __restrict__ out,
                              int Cin, int Cout, int nkc, int nmt, int total) {
    int idx = blockIdx.x * 256 + threadIdx.x;
    if (idx >= total) return;
    int i    = idx & 15;
    int lane = (idx >> 4) & 31;
    int mt   = (idx >> 9) % nmt;
    int t2   = idx / (512 * nmt);
    int kc   = t2 % nkc;
    int r    = t2 / nkc;
    int half = lane >> 4;
    int co   = mt * 16 + (lane & 15);
    int kk   = 16 * half + i;
    int ci   = kc * 32 + kk;
    float v  = (co < Cout && ci < Cin) ? W[((size_t)co * Cin + ci) * 9 + r] : 0.f;
    out[idx] = f2bf(v);
}

// dcn weights: K = (g*9+k)*16 + c ; layout [kc(18)][mt(4)][lane][16]
__global__ void pack_dcn_kernel(const float* __restrict__ W, unsigned short* __restrict__ out) {
    int idx = blockIdx.x * 256 + threadIdx.x;
    if (idx >= 18 * 4 * 512) return;
    int i    = idx & 15;
    int lane = (idx >> 4) & 31;
    int mt   = (idx >> 9) & 3;
    int kc   = idx >> 11;
    int half = lane >> 4;
    int co   = mt * 16 + (lane & 15);
    int kk   = 16 * half + i;
    int K    = kc * 32 + kk;      // 0..575
    int gk   = K >> 4;            // g*9 + k
    int c    = K & 15;
    int g    = gk / 9, k = gk % 9;
    float v  = W[((size_t)co * 64 + (g * 16 + c)) * 9 + k];
    out[idx] = f2bf(v);
}

// ---------------------------------------------------------------------------
// Concat / convert / upsample builder kernels (all write bf16)
// ---------------------------------------------------------------------------
__global__ void cat0_kernel(const float* __restrict__ xa, const float* __restrict__ xb,
                            const float* __restrict__ xc, const float* __restrict__ xd,
                            const float* __restrict__ xe, unsigned short* __restrict__ out) {
    size_t idx = (size_t)blockIdx.x * 256 + threadIdx.x;   // [8][128][HW]
    int pix = (int)(idx & (HWSZ - 1));
    int ch  = (int)((idx >> 14) & 127);
    int n   = (int)(idx >> 21);
    int b = n & 1, br = n >> 1;
    float v;
    if (ch < 64) {
        v = xc[((size_t)b * 64 + ch) * HWSZ + pix];
    } else {
        const float* r = (br == 0) ? xa : (br == 1) ? xb : (br == 2) ? xd : xe;
        v = r[((size_t)b * 64 + (ch - 64)) * HWSZ + pix];
    }
    out[idx] = f2bf(v);
}

__global__ void cat1_kernel(const unsigned short* __restrict__ kpa, const float* __restrict__ xc,
                            unsigned short* __restrict__ out) {
    size_t idx = (size_t)blockIdx.x * 256 + threadIdx.x;   // [8][128][HW]
    int pix = (int)(idx & (HWSZ - 1));
    int ch  = (int)((idx >> 14) & 127);
    int n   = (int)(idx >> 21);
    if (ch < 64) out[idx] = kpa[((size_t)n * 64 + ch) * HWSZ + pix];
    else         out[idx] = f2bf(xc[((size_t)(n & 1) * 64 + (ch - 64)) * HWSZ + pix]);
}

// bilinear 2x upsample, half-pixel centers, edge clamp; src is one 64x64 plane
__device__ __forceinline__ float up2s(const float* __restrict__ src, int y, int x) {
    float sy = 0.5f * y - 0.25f, sx = 0.5f * x - 0.25f;
    float y0f = floorf(sy), x0f = floorf(sx);
    float wy = sy - y0f, wx = sx - x0f;
    int y0 = (int)y0f, x0 = (int)x0f;
    int y0c = min(max(y0, 0), 63), y1c = min(max(y0 + 1, 0), 63);
    int x0c = min(max(x0, 0), 63), x1c = min(max(x0 + 1, 0), 63);
    float a = src[y0c * 64 + x0c], b = src[y0c * 64 + x1c];
    float c = src[y1c * 64 + x0c], d = src[y1c * 64 + x1c];
    return (1.f - wy) * ((1.f - wx) * a + wx * b) + wy * ((1.f - wx) * c + wx * d);
}

__global__ void cat2_kernel(const unsigned short* __restrict__ off1, const float* __restrict__ prevOff,
                            unsigned short* __restrict__ out, size_t total) {
    size_t idx = (size_t)blockIdx.x * 256 + threadIdx.x;   // [8][172][HW]
    if (idx >= total) return;
    int pix = (int)(idx % HWSZ);
    int ch  = (int)((idx / HWSZ) % 172);
    int n   = (int)(idx / ((size_t)172 * HWSZ));
    int x = pix & 127, y = pix >> 7;
    if (ch < 64) out[idx] = off1[((size_t)n * 64 + ch) * HWSZ + pix];
    else         out[idx] = f2bf(up2s(prevOff + ((size_t)n * 108 + (ch - 64)) * 4096, y, x));
}

__global__ void cat3_kernel(const unsigned short* __restrict__ aligned, const float* __restrict__ prevFeat,
                            unsigned short* __restrict__ out) {
    size_t idx = (size_t)blockIdx.x * 256 + threadIdx.x;   // [8][128][HW]
    int pix = (int)(idx & (HWSZ - 1));
    int ch  = (int)((idx >> 14) & 127);
    int n   = (int)(idx >> 21);
    int x = pix & 127, y = pix >> 7;
    if (ch < 64) out[idx] = aligned[((size_t)n * 64 + ch) * HWSZ + pix];
    else         out[idx] = f2bf(up2s(prevFeat + ((size_t)n * 64 + (ch - 64)) * 4096, y, x));
}

__global__ void copy_xc_kernel(const float* __restrict__ xc, float* __restrict__ out) {
    size_t idx = (size_t)blockIdx.x * 256 + threadIdx.x;
    if (idx < (size_t)ALIGN_SZ) out[idx] = xc[idx];
}

// ---------------------------------------------------------------------------
// 3x3 SAME conv as implicit GEMM via v_wmma_f32_16x16x32_bf16.
// grid = N4*H row-blocks; block = 256 (8 waves). A block computes one output
// row (128 px) x ALL Cout tiles: each wave owns 16 pixels and keeps NMT fp32
// accumulators, so the activation slab is staged through LDS exactly once
// (HBM traffic /NMT vs. tiling Cout in the grid). Per tap: one B fragment
// (two contiguous ds_load_b128) feeds NMT back-to-back WMMAs whose A
// fragments are consecutive 32B global loads from the weight pack.
// ---------------------------------------------------------------------------
template <int NMT>
__global__ __launch_bounds__(256) void conv3x3_wmma(
    const unsigned short* __restrict__ X,      // bf16 [N4][Cin][H][W]
    const unsigned short* __restrict__ Apack,  // bf16 [9][nkc][NMT][32][16]
    const float* __restrict__ bias,
    int Cin, int Cout, int nkc, int relu, int mode,
    unsigned short* __restrict__ dstBf, float* __restrict__ dstF32)
{
    __shared__ __align__(16) unsigned short Xs[3 * 136 * 32];
    int rb = blockIdx.x;
    int n = rb >> 7, y = rb & 127;
    int tid = threadIdx.x;
    int lane = tid & 31, wave = tid >> 5;
    int half = lane >> 4, nn = lane & 15;
    int xpix = wave * 16 + nn;

    v8f acc[NMT];
#pragma unroll
    for (int m = 0; m < NMT; ++m) acc[m] = (v8f){0.f, 0.f, 0.f, 0.f, 0.f, 0.f, 0.f, 0.f};

    for (int kc = 0; kc < nkc; ++kc) {
        __syncthreads();
        // ---- stage LDS slab: 32 channels x 3 rows x [-1,135) ----
        for (int task = tid; task < 32 * 3 * 17; task += 256) {
            int s   = task % 17;
            int row = (task / 17) % 3;
            int ci  = task / 51;
            int cig = kc * 32 + ci;
            int yy  = y - 1 + row;
            bool rowok = (cig < Cin) && (yy >= 0) && (yy < HH);
            int yyc  = min(max(yy, 0), HH - 1);
            int cigc = min(cig, Cin - 1);
            const unsigned short* src = X + (((size_t)n * Cin + cigc) * HH + yyc) * WW;
            int x0 = s * 8 - 1;
            unsigned short vals[8];
            if (rowok && x0 >= 0 && x0 + 8 <= WW) {
                uint4 qv = *(const uint4*)(src + x0);
                *(uint4*)vals = qv;
            } else {
#pragma unroll
                for (int e = 0; e < 8; ++e) {
                    int xg = x0 + e;
                    vals[e] = (rowok && xg >= 0 && xg < WW) ? src[xg] : (unsigned short)0;
                }
            }
#pragma unroll
            for (int e = 0; e < 8; ++e) {
                int xx = x0 + 1 + e;                  // 0..135
                Xs[(row * 136 + xx) * 32 + ci] = vals[e];
            }
            if (kc + 1 < nkc)   // warm L2 for next chunk
                __builtin_prefetch(X + (((size_t)n * Cin + min(cig + 32, Cin - 1)) * HH + yyc) * WW, 0, 1);
        }
        __syncthreads();
        // ---- 9 taps: B fragment loaded once, NMT WMMAs each ----
#pragma unroll
        for (int r = 0; r < 9; ++r) {
            int ky = r / 3, kx = r % 3;
            BF16x16 bfr;
            const uint4* bp = (const uint4*)&Xs[((ky * 136) + (xpix + kx)) * 32 + half * 16];
            bfr.q[0] = bp[0]; bfr.q[1] = bp[1];
            const unsigned short* abase =
                Apack + (((size_t)r * nkc + kc) * NMT * 32 + lane) * 16;
#pragma unroll
            for (int mt = 0; mt < NMT; ++mt) {
                BF16x16 afr;
                const uint4* ap = (const uint4*)(abase + (size_t)mt * 512 * 16 / 16 * 16);
                ap = (const uint4*)(abase + (size_t)mt * 32 * 16);
                afr.q[0] = ap[0]; afr.q[1] = ap[1];
                acc[mt] = __builtin_amdgcn_wmma_f32_16x16x32_bf16(false, afr.v, false, bfr.v,
                                                                  (short)0, acc[mt], false, false);
            }
        }
    }

    // ---- store: C/D layout lane=N(pixel), vgpr v -> M = v + 8*half ----
#pragma unroll
    for (int mt = 0; mt < NMT; ++mt) {
#pragma unroll
        for (int v = 0; v < 8; ++v) {
            int co = mt * 16 + v + 8 * half;
            if (co < Cout) {
                float val = acc[mt][v] + bias[co];
                if (relu) val = fmaxf(val, 0.f);
                if (mode == 0) {
                    dstBf[(((size_t)n * Cout + co) * HH + y) * WW + xpix] = f2bf(val);
                } else if (mode == 1) {
                    dstF32[(((size_t)n * Cout + co) * HH + y) * WW + xpix] = val;
                } else {
                    int reg = n >> 1; if (reg >= 2) reg++;   // skip xc slot
                    dstF32[(size_t)reg * ALIGN_SZ + (((size_t)(n & 1) * 64 + co) * HH + y) * WW + xpix] = val;
                }
            }
        }
    }
}

// ---------------------------------------------------------------------------
// Modulated deformable conv v2 as WMMA GEMM: M=64 (4 tiles), K=G*Cg*9=576
// (18 chunks of 32), N=16 pixels/wave. Each lane's half selects one (g,k)
// tap; bilinear coords + sigmoid(mask) computed once, then 16-channel gather
// builds the bf16 B fragment (reused by all 4 M-tiles).
// ---------------------------------------------------------------------------
__global__ __launch_bounds__(256) void deform_wmma_kernel(
    const unsigned short* __restrict__ kpa,   // bf16 [8][64][HW]
    const float* __restrict__ offD,           // fp32 [8][108][HW] (off_feat)
    const unsigned short* __restrict__ pack,  // bf16 [18][4][32][16]
    const float* __restrict__ bias,
    unsigned short* __restrict__ outBf)       // bf16 [8][64][HW]
{
    int tid = threadIdx.x, lane = tid & 31, wave = tid >> 5;
    int half = lane >> 4, nn = lane & 15;
    int tile = blockIdx.x * 8 + wave;
    int p = tile * 16 + nn;
    int n = p >> 14;
    int y = (p >> 7) & 127;
    int x = p & 127;

    v8f acc[4];
#pragma unroll
    for (int m = 0; m < 4; ++m) acc[m] = (v8f){0.f, 0.f, 0.f, 0.f, 0.f, 0.f, 0.f, 0.f};

    for (int kc = 0; kc < 18; ++kc) {
        int gk = kc * 2 + half;          // g*9 + k
        int g = gk / 9, k = gk % 9;
        size_t ob = ((size_t)n * 108) * HWSZ + (size_t)y * WW + x;
        float offy = offD[ob + (size_t)gk * HWSZ];
        float offx = offD[ob + (size_t)(36 + gk) * HWSZ];
        float mrw  = offD[ob + (size_t)(72 + gk) * HWSZ];
        float m = 1.f / (1.f + __expf(-mrw));
        float py = (float)y + (float)(k / 3) - 1.f + offy;
        float px = (float)x + (float)(k % 3) - 1.f + offx;
        float y0f = floorf(py), x0f = floorf(px);
        int y0 = (int)y0f, x0 = (int)x0f;
        float wy = py - y0f, wx = px - x0f;
        int y1 = y0 + 1, x1 = x0 + 1;
        bool vy0 = (y0 >= 0) && (y0 < HH), vy1 = (y1 >= 0) && (y1 < HH);
        bool vx0 = (x0 >= 0) && (x0 < WW), vx1 = (x1 >= 0) && (x1 < WW);
        int y0c = min(max(y0, 0), HH - 1), y1c = min(max(y1, 0), HH - 1);
        int x0c = min(max(x0, 0), WW - 1), x1c = min(max(x1, 0), WW - 1);
        float w00 = (vy0 && vx0) ? (1.f - wy) * (1.f - wx) * m : 0.f;
        float w01 = (vy0 && vx1) ? (1.f - wy) * wx * m : 0.f;
        float w10 = (vy1 && vx0) ? wy * (1.f - wx) * m : 0.f;
        float w11 = (vy1 && vx1) ? wy * wx * m : 0.f;
        int o00 = y0c * WW + x0c, o01 = y0c * WW + x1c;
        int o10 = y1c * WW + x0c, o11 = y1c * WW + x1c;
        const unsigned short* src = kpa + ((size_t)n * 64 + g * 16) * HWSZ;

        BF16x16 bfr;
#pragma unroll
        for (int c = 0; c < 16; ++c) {
            const unsigned short* sc = src + (size_t)c * HWSZ;
            float s = w00 * bf2f(sc[o00]) + w01 * bf2f(sc[o01])
                    + w10 * bf2f(sc[o10]) + w11 * bf2f(sc[o11]);
            bfr.u[c] = f2bf(s);
        }
#pragma unroll
        for (int mtl = 0; mtl < 4; ++mtl) {
            BF16x16 afr;
            const uint4* ap = (const uint4*)(pack + (((size_t)kc * 4 + mtl) * 32 + lane) * 16);
            afr.q[0] = ap[0]; afr.q[1] = ap[1];
            acc[mtl] = __builtin_amdgcn_wmma_f32_16x16x32_bf16(false, afr.v, false, bfr.v,
                                                               (short)0, acc[mtl], false, false);
        }
    }
#pragma unroll
    for (int mtl = 0; mtl < 4; ++mtl) {
#pragma unroll
        for (int v = 0; v < 8; ++v) {
            int co = mtl * 16 + v + 8 * half;
            float val = acc[mtl][v] + bias[co];
            outBf[((size_t)n * 64 + co) * HWSZ + (size_t)y * WW + x] = f2bf(val);
        }
    }
}

// ---------------------------------------------------------------------------
// Host-side orchestration
// ---------------------------------------------------------------------------
extern "C" void kernel_launch(void* const* d_in, const int* in_sizes, int n_in,
                              void* d_out, int out_size, void* d_ws, size_t ws_size,
                              hipStream_t stream) {
    (void)in_sizes; (void)n_in; (void)out_size; (void)ws_size;
    const float* xa = (const float*)d_in[0];
    const float* xb = (const float*)d_in[1];
    const float* xc = (const float*)d_in[2];
    const float* xd = (const float*)d_in[3];
    const float* xe = (const float*)d_in[4];
    const float* prev_off  = (const float*)d_in[5];
    const float* prev_feat = (const float*)d_in[6];
    const float* kpa_w = (const float*)d_in[7];   const float* kpa_b = (const float*)d_in[8];
    const float* oc1_w1 = (const float*)d_in[9];  const float* oc1_b1 = (const float*)d_in[10];
    const float* oc1_w2 = (const float*)d_in[11]; const float* oc1_b2 = (const float*)d_in[12];
    const float* oc2_w1 = (const float*)d_in[13]; const float* oc2_b1 = (const float*)d_in[14];
    const float* oc2_w2 = (const float*)d_in[15]; const float* oc2_b2 = (const float*)d_in[16];
    const float* fuse_w1 = (const float*)d_in[17]; const float* fuse_b1 = (const float*)d_in[18];
    const float* fuse_w2 = (const float*)d_in[19]; const float* fuse_b2 = (const float*)d_in[20];
    const float* dcn_w = (const float*)d_in[21];  const float* dcn_b = (const float*)d_in[22];
    float* out = (float*)d_out;
    char* ws = (char*)d_ws;

    // workspace layout (bytes); P = N4*HW = 131072 pixels
    unsigned short* catA = (unsigned short*)(ws + 0);           // 172*P bf16 (cat0 / cat2)
    unsigned short* catB = (unsigned short*)(ws + 45088768);    // 128*P bf16 (cat1 / cat3)
    unsigned short* kpaO = (unsigned short*)(ws + 78643200);    // 64*P
    unsigned short* hbuf = (unsigned short*)(ws + 95420416);    // 64*P
    unsigned short* xbuf = (unsigned short*)(ws + 112197632);   // 64*P (off1 then aligned)
    unsigned short* pk_kpa  = (unsigned short*)(ws + 128974848);
    unsigned short* pk_oc11 = (unsigned short*)(ws + 129122304);
    unsigned short* pk_oc12 = (unsigned short*)(ws + 129269760);
    unsigned short* pk_oc21 = (unsigned short*)(ws + 129343488);
    unsigned short* pk_oc22 = (unsigned short*)(ws + 129564672);
    unsigned short* pk_f1   = (unsigned short*)(ws + 129693696);
    unsigned short* pk_f2   = (unsigned short*)(ws + 129841152);
    unsigned short* pk_dcn  = (unsigned short*)(ws + 129914880);

    auto pack = [&](const float* w, unsigned short* dst, int Cin, int Cout) {
        int nkc = (Cin + 31) / 32, nmt = (Cout + 15) / 16;
        int total = 9 * nkc * nmt * 512;
        pack_w_kernel<<<(total + 255) / 256, 256, 0, stream>>>(w, dst, Cin, Cout, nkc, nmt, total);
    };
    auto conv = [&](const unsigned short* X, const unsigned short* pk, const float* b,
                    int Cin, int Cout, int relu, int mode, unsigned short* dbf, float* df) {
        int nkc = (Cin + 31) / 32, nmt = (Cout + 15) / 16;
        dim3 grid(N4 * HH, 1, 1);
        if (nmt == 4)
            conv3x3_wmma<4><<<grid, 256, 0, stream>>>(X, pk, b, Cin, Cout, nkc, relu, mode, dbf, df);
        else
            conv3x3_wmma<7><<<grid, 256, 0, stream>>>(X, pk, b, Cin, Cout, nkc, relu, mode, dbf, df);
    };

    // weight packing (tiny)
    pack(kpa_w,  pk_kpa,  128, 64);
    pack(oc1_w1, pk_oc11, 128, 64);
    pack(oc1_w2, pk_oc12, 64,  64);
    pack(oc2_w1, pk_oc21, 172, 64);
    pack(oc2_w2, pk_oc22, 64, 108);
    pack(fuse_w1, pk_f1,  128, 64);
    pack(fuse_w2, pk_f2,  64,  64);
    pack_dcn_kernel<<<144, 256, 0, stream>>>(dcn_w, pk_dcn);

    // kpa
    cat0_kernel<<<65536, 256, 0, stream>>>(xa, xb, xc, xd, xe, catA);
    conv(catA, pk_kpa, kpa_b, 128, 64, 1, 0, kpaO, nullptr);
    // offset_conv_1
    cat1_kernel<<<65536, 256, 0, stream>>>(kpaO, xc, catB);
    conv(catB, pk_oc11, oc1_b1, 128, 64, 1, 0, hbuf, nullptr);
    conv(hbuf, pk_oc12, oc1_b2, 64, 64, 0, 0, xbuf, nullptr);
    // offset_conv_2 (with fused up2 of prev_offset_feat)
    {
        size_t total = (size_t)N4 * 172 * HWSZ;
        cat2_kernel<<<(unsigned)((total + 255) / 256), 256, 0, stream>>>(xbuf, prev_off, catA, total);
    }
    conv(catA, pk_oc21, oc2_b1, 172, 64, 1, 0, hbuf, nullptr);
    conv(hbuf, pk_oc22, oc2_b2, 64, 108, 0, 1, nullptr, out + OFF_BASE);   // off_feat -> d_out
    // deformable conv (reads offsets/mask straight from d_out off region)
    deform_wmma_kernel<<<1024, 256, 0, stream>>>(kpaO, out + OFF_BASE, pk_dcn, dcn_b, xbuf);
    // fuse (with fused up2 of prev_feat)
    cat3_kernel<<<65536, 256, 0, stream>>>(xbuf, prev_feat, catB);
    conv(catB, pk_f1, fuse_b1, 128, 64, 1, 0, hbuf, nullptr);
    conv(hbuf, pk_f2, fuse_b2, 64, 64, 0, 2, nullptr, out);                // aligned -> d_out
    // xc passthrough
    copy_xc_kernel<<<8192, 256, 0, stream>>>(xc, out + 2 * (size_t)ALIGN_SZ);
}
